// CustomBert6_21354577396205
// MI455X (gfx1250) — matrix operand order repaired and checked by reading, not compiled, not run on hardware
//
#include <hip/hip_runtime.h>

#define BB 32
#define SQL 400
#define HD 768
#define NHEAD 12
#define DHEAD 64
#define FFD 3072
#define MAXS_ 512
#define MAXQ_ 271
#define MAXA_ 241
#define NFEAT 74
#define NHID_ 256
#define NOUT_ 30
#define NEGBIG -1000000000.0f

#if defined(__has_builtin)
#if __has_builtin(__builtin_amdgcn_global_load_async_to_lds_b128) && \
    __has_builtin(__builtin_amdgcn_s_wait_asynccnt)
#define HAVE_ASYNC_LDS 1
#endif
#endif

typedef __attribute__((ext_vector_type(16))) _Float16 v16h;
typedef __attribute__((ext_vector_type(8)))  _Float16 v8h;
typedef __attribute__((ext_vector_type(8)))  float    v8f;

static __device__ __forceinline__ int laneid() { return threadIdx.x & 31; }

// A-fragment: 16x32 f16 tile (row-major, leading dim ld).
// lane L -> M=L%16, half=L/16; elems 0..7 = K 8h..8h+7, elems 8..15 = K 16+8h..23+8h.
static __device__ __forceinline__ v16h frag_A(const _Float16* t, int ld) {
  int l = laneid();
  const _Float16* r = t + (l & 15) * ld + ((l >> 4) << 3);
  v8h lo = *(const v8h*)(r);
  v8h hi = *(const v8h*)(r + 16);
  v16h o;
#pragma unroll
  for (int i = 0; i < 8; ++i) { o[i] = lo[i]; o[i + 8] = hi[i]; }
  return o;
}

// B-fragment from TRANSPOSED tile tT[n][k]: lane L -> N=L%16; elems 0..7 = K 8h..8h+7, elems 8..15 = K 16+8h..23+8h.
static __device__ __forceinline__ v16h frag_B(const _Float16* tT, int ld) {
  int l = laneid();
  const _Float16* r = tT + (l & 15) * ld + ((l >> 4) << 3);
  v8h lo = *(const v8h*)(r);
  v8h hi = *(const v8h*)(r + 16);
  v16h o;
#pragma unroll
  for (int i = 0; i < 8; ++i) { o[i] = lo[i]; o[i + 8] = hi[i]; }
  return o;
}

static __device__ __forceinline__ v8f wmma_f16(v16h a, v16h b, v8f c) {
  return __builtin_amdgcn_wmma_f32_16x16x32_f16(false, a, false, b, (short)0, c, false, false);
}

#if HAVE_ASYNC_LDS
typedef int v4i_ __attribute__((vector_size(16)));
typedef __attribute__((address_space(1))) v4i_* gv4p;
typedef __attribute__((address_space(3))) v4i_* lv4p;
static __device__ __forceinline__ void async_cp16(const _Float16* g, _Float16* l) {
  __builtin_amdgcn_global_load_async_to_lds_b128((gv4p)g, (lv4p)l, 0, 0);
}
#endif

// ---------- weight transpose+convert: WT[n*K+k] = (f16)W[k*N+n]; K,N multiples of 64 ----------
__global__ __launch_bounds__(256)
void wconv_kernel(const float* __restrict__ W, _Float16* __restrict__ WT, int K, int N) {
  __shared__ _Float16 t[64][65];
  int k0 = blockIdx.x * 64, n0 = blockIdx.y * 64;
  for (int idx = threadIdx.x; idx < 64 * 64; idx += 256) {
    int i = idx >> 6, j = idx & 63;               // i along k, j along n (coalesced)
    t[i][j] = (_Float16)W[(size_t)(k0 + i) * N + (n0 + j)];
  }
  __syncthreads();
  for (int idx = threadIdx.x; idx < 64 * 64; idx += 256) {
    int j = idx >> 6, i = idx & 63;               // write row n=j, contiguous along k
    WT[(size_t)(n0 + j) * K + (k0 + i)] = t[i][j];
  }
}

// ---------- f32 -> f16 convert (n multiple of 1024) ----------
__global__ __launch_bounds__(256)
void conv16_kernel(const float* __restrict__ X, _Float16* __restrict__ Y, size_t n) {
  size_t i = ((size_t)blockIdx.x * 256 + threadIdx.x) * 4;
  if (i + 3 < n) {
    float4 v = *(const float4*)(X + i);
    _Float16 h0 = (_Float16)v.x, h1 = (_Float16)v.y, h2 = (_Float16)v.z, h3 = (_Float16)v.w;
    Y[i] = h0; Y[i + 1] = h1; Y[i + 2] = h2; Y[i + 3] = h3;
  }
}

// ---------- fast GEMM (f16 in, f32 or f16 out): M%128==0, N%64==0, K%32==0 ----------
// C = act(A[M,K] @ BT[N,K]^T + bias). Double-buffered LDS ping-pong; async LDS copies when available.
__global__ __launch_bounds__(128)
void gemm16_kernel(const _Float16* __restrict__ A, const _Float16* __restrict__ BT,
                   const float* __restrict__ bias, float* __restrict__ C32,
                   _Float16* __restrict__ C16, int M, int N, int K, int act) {
  __shared__ __align__(32) _Float16 As[2][128 * 32];
  __shared__ __align__(32) _Float16 Bs[2][64 * 32];   // [n][k]
  int n0 = blockIdx.x * 64, m0 = blockIdx.y * 128;
  int tid = threadIdx.x, w = tid >> 5;
  v8f acc[2][4];
#pragma unroll
  for (int s = 0; s < 2; ++s)
#pragma unroll
    for (int n = 0; n < 4; ++n)
#pragma unroll
      for (int i = 0; i < 8; ++i) acc[s][n][i] = 0.f;

  int r0 = tid >> 1, kb = (tid & 1) << 4;
  const _Float16* pa0 = A + (size_t)(m0 + r0) * K + kb;
  const _Float16* pa1 = A + (size_t)(m0 + 64 + r0) * K + kb;
  const _Float16* pb  = BT + (size_t)(n0 + r0) * K + kb;
  int nk = K >> 5;

#if HAVE_ASYNC_LDS
  // prologue: async-stage tile 0 into buffer 0 (3 x b128 per lane)
  async_cp16(pa0, As[0] + r0 * 32 + kb);
  async_cp16(pa1, As[0] + (64 + r0) * 32 + kb);
  async_cp16(pb,  Bs[0] + r0 * 32 + kb);
  for (int kt = 0; kt < nk; ++kt) {
    int buf = kt & 1;
    __builtin_amdgcn_s_wait_asynccnt(0);             // this wave's copies into buf done
    __syncthreads();                                 // everyone's copies done; buf^1 reads retired
    if (kt + 1 < nk) {                               // DMA next tile while WMMAs run
      size_t o = (size_t)(kt + 1) << 5;
      int nb = buf ^ 1;
      async_cp16(pa0 + o, As[nb] + r0 * 32 + kb);
      async_cp16(pa1 + o, As[nb] + (64 + r0) * 32 + kb);
      async_cp16(pb + o,  Bs[nb] + r0 * 32 + kb);
      if (kt + 2 < nk) {
        __builtin_prefetch(pa0 + o + 32, 0, 1);
        __builtin_prefetch(pb + o + 32, 0, 1);
      }
    }
    v16h a0 = frag_A(As[buf] + (w * 32) * 32, 32);
    v16h a1 = frag_A(As[buf] + (w * 32 + 16) * 32, 32);
#pragma unroll
    for (int n = 0; n < 4; ++n) {
      v16h b = frag_B(Bs[buf] + n * 16 * 32, 32);
      acc[0][n] = wmma_f16(a0, b, acc[0][n]);
      acc[1][n] = wmma_f16(a1, b, acc[1][n]);
    }
  }
#else
  // prologue: stage tile 0 into buffer 0 through registers
  uint4 ra0 = *(const uint4*)(pa0);
  uint4 ra1 = *(const uint4*)(pa1);
  uint4 rb  = *(const uint4*)(pb);
  *(uint4*)(As[0] + r0 * 32 + kb)        = ra0;
  *(uint4*)(As[0] + (64 + r0) * 32 + kb) = ra1;
  *(uint4*)(Bs[0] + r0 * 32 + kb)        = rb;
  for (int kt = 0; kt < nk; ++kt) {
    int buf = kt & 1;
    __syncthreads();                                 // buf ready; buf^1 free to overwrite
    if (kt + 1 < nk) {                               // issue next tile's global loads early
      size_t o = (size_t)(kt + 1) << 5;
      ra0 = *(const uint4*)(pa0 + o);
      ra1 = *(const uint4*)(pa1 + o);
      rb  = *(const uint4*)(pb + o);
      if (kt + 2 < nk) {
        __builtin_prefetch(pa0 + o + 32, 0, 1);
        __builtin_prefetch(pb + o + 32, 0, 1);
      }
    }
    v16h a0 = frag_A(As[buf] + (w * 32) * 32, 32);
    v16h a1 = frag_A(As[buf] + (w * 32 + 16) * 32, 32);
#pragma unroll
    for (int n = 0; n < 4; ++n) {
      v16h b = frag_B(Bs[buf] + n * 16 * 32, 32);
      acc[0][n] = wmma_f16(a0, b, acc[0][n]);
      acc[1][n] = wmma_f16(a1, b, acc[1][n]);
    }
    if (kt + 1 < nk) {                               // spill next tile into other buffer
      int nb = buf ^ 1;
      *(uint4*)(As[nb] + r0 * 32 + kb)        = ra0;
      *(uint4*)(As[nb] + (64 + r0) * 32 + kb) = ra1;
      *(uint4*)(Bs[nb] + r0 * 32 + kb)        = rb;
    }
  }
#endif

  int col = laneid() & 15, half = laneid() >> 4;
#pragma unroll
  for (int sub = 0; sub < 2; ++sub)
#pragma unroll
    for (int n = 0; n < 4; ++n) {
      int gn = n0 + n * 16 + col;
      float bv = bias[gn];
#pragma unroll
      for (int r = 0; r < 8; ++r) {
        int gm = m0 + w * 32 + sub * 16 + half * 8 + r;
        float v = acc[sub][n][r] + bv;
        if (act == 1) v = 0.5f * v * (1.f + erff(v * 0.70710678118f));
        else if (act == 2) v = fmaxf(v, 0.f);
        if (C32) C32[(size_t)gm * N + gn] = v;
        else     C16[(size_t)gm * N + gn] = (_Float16)v;
      }
    }
}

// ---------- guarded GEMM for small head layers (f32 in/out, clamped unconditional loads) ----------
__global__ __launch_bounds__(128)
void gemm_edge_kernel(const float* __restrict__ A, const float* __restrict__ Bw,
                      const float* __restrict__ bias, float* __restrict__ C,
                      int M, int N, int K, int ldc, int act) {
  __shared__ __align__(32) _Float16 As[64 * 32];
  __shared__ __align__(32) _Float16 Bs[64 * 32];
  int n0 = blockIdx.x * 64, m0 = blockIdx.y * 64;
  int tid = threadIdx.x, w = tid >> 5;
  v8f acc[4];
#pragma unroll
  for (int n = 0; n < 4; ++n)
#pragma unroll
    for (int i = 0; i < 8; ++i) acc[n][i] = 0.f;

  for (int k0 = 0; k0 < K; k0 += 32) {
    __syncthreads();
    {
      int row = tid >> 1, kb = (tid & 1) << 4;
      int gm = m0 + row, gmc = gm < M ? gm : M - 1;
      const float* ap = A + (size_t)gmc * K;
#pragma unroll
      for (int i = 0; i < 16; ++i) {
        int gk = k0 + kb + i, gkc = gk < K ? gk : K - 1;
        float v = ap[gkc];
        As[row * 32 + kb + i] = (_Float16)((gm < M && gk < K) ? v : 0.f);
      }
    }
    {
      int n = tid & 63, kb = (tid >> 6) << 4;
      int gn = n0 + n, gnc = gn < N ? gn : N - 1;
#pragma unroll
      for (int i = 0; i < 16; ++i) {
        int gk = k0 + kb + i, gkc = gk < K ? gk : K - 1;
        float v = Bw[(size_t)gkc * N + gnc];
        Bs[n * 32 + kb + i] = (_Float16)((gk < K && gn < N) ? v : 0.f);
      }
    }
    __syncthreads();
    v16h a = frag_A(As + w * 16 * 32, 32);
#pragma unroll
    for (int n = 0; n < 4; ++n) {
      v16h b = frag_B(Bs + n * 16 * 32, 32);
      acc[n] = wmma_f16(a, b, acc[n]);
    }
  }
  int col = laneid() & 15, half = laneid() >> 4;
#pragma unroll
  for (int n = 0; n < 4; ++n) {
    int gn = n0 + n * 16 + col;
    if (gn >= N) continue;
    float bv = bias[gn];
#pragma unroll
    for (int r = 0; r < 8; ++r) {
      int gm = m0 + w * 16 + half * 8 + r;
      if (gm >= M) continue;
      float v = acc[n][r] + bv;
      if (act == 1) v = 0.5f * v * (1.f + erff(v * 0.70710678118f));
      else if (act == 2) v = fmaxf(v, 0.f);
      C[(size_t)gm * ldc + gn] = v;
    }
  }
}

// ---------- flash attention (f16 in/out): per (b, h, 64-query tile) ----------
__global__ __launch_bounds__(128)
void attn_kernel(const _Float16* __restrict__ Q, const _Float16* __restrict__ Kk,
                 const _Float16* __restrict__ V, const int* __restrict__ mask,
                 _Float16* __restrict__ O, int S) {
  __shared__ __align__(32) _Float16 Qs[64 * 64];    // [qrow][d], pre-scaled
  __shared__ __align__(32) _Float16 Ks[32 * 64];    // [key][d]  == B^T tile for scores
  __shared__ __align__(32) _Float16 Vs[64 * 32];    // [d][key]  == B^T tile for O
  __shared__ __align__(32) _Float16 Ps[4][16 * 32];
  int qb = blockIdx.x * 64;
  int h = blockIdx.y, b = blockIdx.z;
  int tid = threadIdx.x, w = tid >> 5, l = laneid();
  int col = l & 15, half = l >> 4;
  size_t baseBH = ((size_t)b * S) * (NHEAD * DHEAD) + (size_t)h * DHEAD;

#pragma unroll
  for (int it = 0; it < 4; ++it) {                  // 4096 halves, 8 per thread per it
    int c = tid + it * 128;
    int row = c >> 3, kb = (c & 7) << 3;
    int s = qb + row, scl = s < S ? s : S - 1;
    v8h hv = *(const v8h*)(Q + baseBH + (size_t)scl * (NHEAD * DHEAD) + kb);
    float zf = (s < S) ? 0.125f : 0.f;
#pragma unroll
    for (int i = 0; i < 8; ++i) hv[i] = (_Float16)((float)hv[i] * zf);
    *(v8h*)(Qs + row * 64 + kb) = hv;
  }
  __syncthreads();
  v16h aq0 = frag_A(Qs + w * 16 * 64, 64);
  v16h aq1 = frag_A(Qs + w * 16 * 64 + 32, 64);

  float mrun[8], lrun[8];
  v8f Oa[4];
#pragma unroll
  for (int r = 0; r < 8; ++r) { mrun[r] = -1e30f; lrun[r] = 0.f; }
#pragma unroll
  for (int n = 0; n < 4; ++n)
#pragma unroll
    for (int i = 0; i < 8; ++i) Oa[n][i] = 0.f;

  int nkt = (S + 31) / 32;
  for (int kt = 0; kt < nkt; ++kt) {
    int ks = kt * 32;
    __syncthreads();
#pragma unroll
    for (int it = 0; it < 2; ++it) {                // 2048 halves, 8 per thread per it
      int c = tid + it * 128;
      int key = c >> 3, kb = (c & 7) << 3;
      int s = ks + key, scl = s < S ? s : S - 1;
      size_t g = baseBH + (size_t)scl * (NHEAD * DHEAD) + kb;
      v8h kv = *(const v8h*)(Kk + g);
      v8h vv = *(const v8h*)(V + g);
      if (s >= S) {
#pragma unroll
        for (int i = 0; i < 8; ++i) { kv[i] = (_Float16)0.f; vv[i] = (_Float16)0.f; }
      }
      *(v8h*)(Ks + key * 64 + kb) = kv;
#pragma unroll
      for (int i = 0; i < 8; ++i) Vs[(kb + i) * 32 + key] = vv[i];
    }
    __syncthreads();

    v8f s0, s1;
#pragma unroll
    for (int i = 0; i < 8; ++i) { s0[i] = 0.f; s1[i] = 0.f; }
    s0 = wmma_f16(aq0, frag_B(Ks, 64), s0);
    s0 = wmma_f16(aq1, frag_B(Ks + 32, 64), s0);
    s1 = wmma_f16(aq0, frag_B(Ks + 16 * 64, 64), s1);
    s1 = wmma_f16(aq1, frag_B(Ks + 16 * 64 + 32, 64), s1);

    int k0i = ks + col, k1i = ks + 16 + col;
    int k0c = k0i < S ? k0i : S - 1, k1c = k1i < S ? k1i : S - 1;
    bool v0 = (k0i < S) && (mask[(size_t)b * S + k0c] != 0);
    bool v1 = (k1i < S) && (mask[(size_t)b * S + k1c] != 0);

    float p0[8], p1[8], sc[8];
#pragma unroll
    for (int r = 0; r < 8; ++r) {
      float a0 = v0 ? s0[r] : NEGBIG;
      float a1 = v1 ? s1[r] : NEGBIG;
      float tm = fmaxf(a0, a1);
#pragma unroll
      for (int x = 1; x < 16; x <<= 1) tm = fmaxf(tm, __shfl_xor(tm, x, 32));
      float mn = fmaxf(mrun[r], tm);
      float scale = expf(mrun[r] - mn);
      float e0 = v0 ? expf(s0[r] - mn) : 0.f;
      float e1 = v1 ? expf(s1[r] - mn) : 0.f;
      float rs = e0 + e1;
#pragma unroll
      for (int x = 1; x < 16; x <<= 1) rs += __shfl_xor(rs, x, 32);
      lrun[r] = lrun[r] * scale + rs;
      mrun[r] = mn;
      sc[r] = scale; p0[r] = e0; p1[r] = e1;
    }
#pragma unroll
    for (int n = 0; n < 4; ++n)
#pragma unroll
      for (int r = 0; r < 8; ++r) Oa[n][r] *= sc[r];

#pragma unroll
    for (int r = 0; r < 8; ++r) {
      Ps[w][(half * 8 + r) * 32 + col]      = (_Float16)p0[r];
      Ps[w][(half * 8 + r) * 32 + 16 + col] = (_Float16)p1[r];
    }
    __syncthreads();
    v16h ap = frag_A(Ps[w], 32);
#pragma unroll
    for (int n = 0; n < 4; ++n) Oa[n] = wmma_f16(ap, frag_B(Vs + n * 16 * 32, 32), Oa[n]);
  }

#pragma unroll
  for (int r = 0; r < 8; ++r) {
    int s = qb + w * 16 + half * 8 + r;
    if (s >= S) continue;
    float inv = 1.f / fmaxf(lrun[r], 1e-20f);
    size_t g = baseBH + (size_t)s * (NHEAD * DHEAD);
#pragma unroll
    for (int n = 0; n < 4; ++n) O[g + n * 16 + col] = (_Float16)(Oa[n][r] * inv);
  }
}

// ---------------- reductions / LN / embed / misc -------------
static __device__ __forceinline__ float block_sum256(float v, float* sh) {
#pragma unroll
  for (int o = 16; o > 0; o >>= 1) v += __shfl_down(v, o, 32);
  int lane = threadIdx.x & 31, wid = threadIdx.x >> 5;
  if (lane == 0) sh[wid] = v;
  __syncthreads();
  if (wid == 0) {
    float r = (lane < 8) ? sh[lane] : 0.f;
#pragma unroll
    for (int o = 4; o > 0; o >>= 1) r += __shfl_down(r, o, 32);
    if (lane == 0) sh[0] = r;
  }
  __syncthreads();
  float r = sh[0];
  __syncthreads();
  return r;
}

__global__ __launch_bounds__(256)
void embed_kernel(const int* __restrict__ ids, const float* __restrict__ word,
                  const float* __restrict__ pos, const float* __restrict__ g,
                  const float* __restrict__ be, float* __restrict__ X, int S) {
  __shared__ float sh[8];
  int m = blockIdx.x;
  int s = m % S;
  int id = ids[m];
  float v[3];
#pragma unroll
  for (int i = 0; i < 3; ++i) {
    int c = threadIdx.x + i * 256;
    v[i] = word[(size_t)id * HD + c] + pos[(size_t)s * HD + c];
  }
  float mean = block_sum256(v[0] + v[1] + v[2], sh) * (1.f / HD);
  float q = 0.f;
#pragma unroll
  for (int i = 0; i < 3; ++i) { float d = v[i] - mean; q += d * d; }
  float var = block_sum256(q, sh) * (1.f / HD);
  float inv = rsqrtf(var + 1e-12f);
#pragma unroll
  for (int i = 0; i < 3; ++i) {
    int c = threadIdx.x + i * 256;
    X[(size_t)m * HD + c] = (v[i] - mean) * inv * g[c] + be[c];
  }
}

__global__ __launch_bounds__(256)
void add_ln_kernel(const float* __restrict__ Ain, const float* __restrict__ R,
                   const float* __restrict__ g, const float* __restrict__ be,
                   float* __restrict__ Y) {
  __shared__ float sh[8];
  size_t m = blockIdx.x;
  float v[3];
#pragma unroll
  for (int i = 0; i < 3; ++i) {
    int c = threadIdx.x + i * 256;
    v[i] = Ain[m * HD + c] + R[m * HD + c];
  }
  float mean = block_sum256(v[0] + v[1] + v[2], sh) * (1.f / HD);
  float q = 0.f;
#pragma unroll
  for (int i = 0; i < 3; ++i) { float d = v[i] - mean; q += d * d; }
  float var = block_sum256(q, sh) * (1.f / HD);
  float inv = rsqrtf(var + 1e-12f);
#pragma unroll
  for (int i = 0; i < 3; ++i) {
    int c = threadIdx.x + i * 256;
    Y[m * HD + c] = (v[i] - mean) * inv * g[c] + be[c];
  }
}

__global__ __launch_bounds__(256)
void counts_kernel(const int* __restrict__ q_ids, const int* __restrict__ a_ids,
                   int* __restrict__ nq, int* __restrict__ na) {
  __shared__ float sh[8];
  int b = blockIdx.x;
  float cq = 0.f, ca = 0.f;
  for (int s = threadIdx.x; s < SQL; s += 256) {
    cq += (q_ids[(size_t)b * SQL + s] > 0) ? 1.f : 0.f;
    ca += (a_ids[(size_t)b * SQL + s] > 0) ? 1.f : 0.f;
  }
  int tq = (int)(block_sum256(cq, sh) + 0.5f);
  int ta = (int)(block_sum256(ca, sh) + 0.5f);
  if (threadIdx.x == 0) {
    int cut = tq + ta - MAXS_;
    int q2 = tq, a2 = ta;
    if (cut > 0) {
      if (tq > MAXQ_) { int t = tq - cut; q2 = (t > MAXQ_) ? t : MAXQ_; }
      if (ta > MAXA_) { int t = ta - cut; a2 = (t > MAXA_) ? t : MAXA_; }
    }
    nq[b] = q2; na[b] = a2;
  }
}

__global__ __launch_bounds__(256)
void concat_kernel(const float* __restrict__ qx, const float* __restrict__ ax,
                   const int* __restrict__ nq, const int* __restrict__ na,
                   float* __restrict__ xc, int* __restrict__ cmask) {
  int j = blockIdx.x, b = blockIdx.y;
  int q2 = nq[b], a2 = na[b];
  const float* src = nullptr;
  if (j < q2) src = qx + ((size_t)b * SQL + j) * HD;
  else if (j < q2 + a2) src = ax + ((size_t)b * SQL + (j - q2)) * HD;
  float* dst = xc + ((size_t)b * MAXS_ + j) * HD;
  for (int c = threadIdx.x; c < HD; c += 256) dst[c] = src ? src[c] : 0.f;
  if (threadIdx.x == 0) cmask[(size_t)b * MAXS_ + j] = (j < q2 + a2) ? 1 : 0;
}

__global__ __launch_bounds__(256)
void pool_kernel(const float* __restrict__ xc, const int* __restrict__ nq,
                 const int* __restrict__ na, float* __restrict__ xq, float* __restrict__ xa) {
  int b = blockIdx.x;
  int q2 = nq[b], a2 = na[b];
  for (int c = threadIdx.x; c < HD; c += 256) {
    float sq = 0.f, sa = 0.f;
    const float* xb = xc + (size_t)b * MAXS_ * HD + c;
    for (int j = 0; j < q2; ++j) sq += xb[(size_t)j * HD];
    for (int j = q2; j < q2 + a2; ++j) sa += xb[(size_t)j * HD];
    xq[(size_t)b * HD + c] = sq / (float)q2;
    xa[(size_t)b * HD + c] = sa / (float)a2;
  }
}

// ---------------- host-side orchestration -------------
static inline void gemm16(const _Float16* A, const _Float16* BT, const float* bias,
                          float* C32, _Float16* C16, int M, int N, int K, int act,
                          hipStream_t st) {
  dim3 g(N / 64, M / 128);
  gemm16_kernel<<<g, 128, 0, st>>>(A, BT, bias, C32, C16, M, N, K, act);
}

static inline void gemm_edge(const float* A, const float* Bw, const float* bias,
                             float* C, int M, int N, int K, int ldc, int act,
                             hipStream_t st) {
  dim3 g((N + 63) / 64, (M + 63) / 64);
  gemm_edge_kernel<<<g, 128, 0, st>>>(A, Bw, bias, C, M, N, K, ldc, act);
}

static void run_block(float* x, _Float16* xh, const int* mask, int S,
                      void* const* d_in, int base, const _Float16* const* WT,
                      _Float16* t0h, _Float16* t1h, _Float16* t2h, _Float16* t3h,
                      _Float16* t4h, float* t5, hipStream_t st) {
  // params: wq,bq,wk,bk,wv,bv,wo,bo,ln1_g,ln1_b,w1,b1,w2,b2,ln2_g,ln2_b
  const float* P[16];
  for (int i = 0; i < 16; ++i) P[i] = (const float*)d_in[base + i];
  int M = BB * S;
  size_t nel = (size_t)M * HD;
  conv16_kernel<<<nel / 1024, 256, 0, st>>>(x, xh, nel);
  gemm16(xh, WT[0], P[1], nullptr, t0h, M, HD, HD, 0, st);            // Q
  gemm16(xh, WT[1], P[3], nullptr, t1h, M, HD, HD, 0, st);            // K
  gemm16(xh, WT[2], P[5], nullptr, t2h, M, HD, HD, 0, st);            // V
  dim3 ga((S + 63) / 64, NHEAD, BB);
  attn_kernel<<<ga, 128, 0, st>>>(t0h, t1h, t2h, mask, t3h, S);
  gemm16(t3h, WT[3], P[7], t5, nullptr, M, HD, HD, 0, st);            // O proj
  add_ln_kernel<<<M, 256, 0, st>>>(t5, x, P[8], P[9], x);             // LN1
  conv16_kernel<<<nel / 1024, 256, 0, st>>>(x, xh, nel);
  const int CH = 2048;
  for (int m0 = 0; m0 < M; m0 += CH) {
    int Mc = (M - m0 < CH) ? (M - m0) : CH;
    gemm16(xh + (size_t)m0 * HD, WT[4], P[11], nullptr, t4h, Mc, FFD, HD, 1, st);  // FFN1+GELU
    gemm16(t4h, WT[5], P[13], t5 + (size_t)m0 * HD, nullptr, Mc, HD, FFD, 0, st);  // FFN2
  }
  add_ln_kernel<<<M, 256, 0, st>>>(t5, x, P[14], P[15], x);           // LN2
}

extern "C" void kernel_launch(void* const* d_in, const int* in_sizes, int n_in,
                              void* d_out, int out_size, void* d_ws, size_t ws_size,
                              hipStream_t stream) {
  (void)in_sizes; (void)n_in; (void)out_size; (void)ws_size;
  // Input order (recursive insertion order of setup_inputs()):
  // 0 x_feats, 1 q_ids, 2 a_ids, 3..6 q_emb, 7..10 a_emb,
  // 11..58 q_blocks, 59..106 a_blocks, 107..154 blocks, 155..162 head
  const float* x_feats = (const float*)d_in[0];
  const int*   q_ids   = (const int*)d_in[1];
  const int*   a_ids   = (const int*)d_in[2];

  float* w = (float*)d_ws;
  size_t off = 0;
  float* qx = w + off; off += (size_t)BB * SQL * HD;
  float* ax = w + off; off += (size_t)BB * SQL * HD;
  float* xc = w + off; off += (size_t)BB * MAXS_ * HD;
  float* t5 = w + off; off += (size_t)BB * MAXS_ * HD;
  float* xqp  = w + off; off += (size_t)BB * HD;
  float* xap  = w + off; off += (size_t)BB * HD;
  float* hbuf = w + off; off += (size_t)BB * HD;
  // f16 regions
  _Float16* hbase = (_Float16*)(w + off);
  size_t hoff = 0;
  const size_t THS = (size_t)BB * MAXS_ * HD;          // max token-major f16 buffer
  _Float16* xh  = hbase + hoff; hoff += THS;
  _Float16* t0h = hbase + hoff; hoff += THS;
  _Float16* t1h = hbase + hoff; hoff += THS;
  _Float16* t2h = hbase + hoff; hoff += THS;
  _Float16* t3h = hbase + hoff; hoff += THS;
  _Float16* t4h = hbase + hoff; hoff += (size_t)2048 * FFD;
  // transposed f16 weights
  const int bases[9] = {11, 27, 43, 59, 75, 91, 107, 123, 139};
  const int widx[6]  = {0, 2, 4, 6, 10, 12};
  const int wkk[6]   = {HD, HD, HD, HD, HD, FFD};
  const int wnn[6]   = {HD, HD, HD, HD, FFD, HD};
  const _Float16* WTs[9][6];
  for (int blk = 0; blk < 9; ++blk)
    for (int j = 0; j < 6; ++j) {
      int K = wkk[j], N = wnn[j];
      _Float16* dst = hbase + hoff; hoff += (size_t)K * N;
      wconv_kernel<<<dim3(K / 64, N / 64), 256, 0, stream>>>(
          (const float*)d_in[bases[blk] + widx[j]], dst, K, N);
      WTs[blk][j] = dst;
    }
  int* ibase = (int*)(hbase + ((hoff + 1) & ~(size_t)1));
  int* cmask = ibase;
  int* nq = ibase + BB * MAXS_;
  int* na = nq + BB;

  embed_kernel<<<BB * SQL, 256, 0, stream>>>(q_ids, (const float*)d_in[3], (const float*)d_in[4],
                                             (const float*)d_in[5], (const float*)d_in[6], qx, SQL);
  embed_kernel<<<BB * SQL, 256, 0, stream>>>(a_ids, (const float*)d_in[7], (const float*)d_in[8],
                                             (const float*)d_in[9], (const float*)d_in[10], ax, SQL);
  for (int i = 0; i < 3; ++i) {
    run_block(qx, xh, q_ids, SQL, d_in, 11 + i * 16, WTs[i],     t0h, t1h, t2h, t3h, t4h, t5, stream);
    run_block(ax, xh, a_ids, SQL, d_in, 59 + i * 16, WTs[3 + i], t0h, t1h, t2h, t3h, t4h, t5, stream);
  }
  counts_kernel<<<BB, 256, 0, stream>>>(q_ids, a_ids, nq, na);
  dim3 gc(MAXS_, BB);
  concat_kernel<<<gc, 256, 0, stream>>>(qx, ax, nq, na, xc, cmask);
  for (int i = 0; i < 3; ++i)
    run_block(xc, xh, cmask, MAXS_, d_in, 107 + i * 16, WTs[6 + i], t0h, t1h, t2h, t3h, t4h, t5, stream);
  pool_kernel<<<BB, 256, 0, stream>>>(xc, nq, na, xqp, xap);

  // head: three ReLU projections into hbuf[32, 768] (ldc=768), then final GEMM
  gemm_edge(x_feats, (const float*)d_in[155], (const float*)d_in[156], hbuf,           BB, NHID_, NFEAT, 3 * NHID_, 2, stream);
  gemm_edge(xqp,     (const float*)d_in[157], (const float*)d_in[158], hbuf + NHID_,   BB, NHID_, HD,    3 * NHID_, 2, stream);
  gemm_edge(xap,     (const float*)d_in[159], (const float*)d_in[160], hbuf + 2*NHID_, BB, NHID_, HD,    3 * NHID_, 2, stream);
  gemm_edge(hbuf,    (const float*)d_in[161], (const float*)d_in[162], (float*)d_out,  BB, NOUT_, 3 * NHID_, NOUT_, 0, stream);
}
